// MambaFiLM_16011638079759
// MI455X (gfx1250) — matrix operand order repaired and checked by reading, not compiled, not run on hardware
//
#include <hip/hip_runtime.h>
#include <math.h>

// ---------------- problem constants (from reference setup_inputs) ------------
constexpr int kB = 8, kT = 2048, kD = 1024, kS = 64, kL = 4, kA = 3;
constexpr int kM = kB * kT;          // 16384 rows
constexpr int kNC = 32, kCH = 64;    // scan chunks: 32 chunks of 64 steps

typedef _Float16 half_t;
typedef __attribute__((ext_vector_type(16))) _Float16 v16h;
typedef __attribute__((ext_vector_type(8)))  float    v8f;

union FragU { uint4 u[2]; v16h v; };

__device__ __forceinline__ float softplusf(float x) {
    return (x > 15.0f) ? x : log1pf(__expf(x));
}

// ---------------- small elementwise kernels ---------------------------------
__global__ void cvt_f16_kernel(const float* __restrict__ src, half_t* __restrict__ dst, int n) {
    int i = blockIdx.x * 256 + threadIdx.x;
    if (i < n) dst[i] = (half_t)src[i];
}

// pack WB/WC (L,64,D) each into WBCh (L,128,D) f16
__global__ void pack_wbc_kernel(const float* __restrict__ WB, const float* __restrict__ WC,
                                half_t* __restrict__ WBCh) {
    int i = blockIdx.x * 256 + threadIdx.x;            // over L*128*D
    int l = i / (128 * kD);
    int r = (i / kD) & 127;
    int d = i & (kD - 1);
    float v = (r < 64) ? WB[((size_t)l * 64 + r) * kD + d]
                       : WC[((size_t)l * 64 + (r - 64)) * kD + d];
    WBCh[i] = (half_t)v;
}

__global__ void pack_bias_kernel(const float* __restrict__ bB, const float* __restrict__ bC,
                                 float* __restrict__ bBC) {
    int i = threadIdx.x;                                // 512 = L*128
    int l = i >> 7, r = i & 127;
    bBC[i] = (r < 64) ? bB[l * 64 + r] : bC[l * 64 + (r - 64)];
}

// sig[l][s] = sigmoid(mean_d A[l,d,s]);  grid = L*S blocks of 256 threads
__global__ void sig_kernel(const float* __restrict__ Am, float* __restrict__ sig) {
    __shared__ float red[256];
    int l = blockIdx.x >> 6, s = blockIdx.x & 63, tid = threadIdx.x;
    float p = 0.f;
    for (int d = tid; d < kD; d += 256) p += Am[((size_t)l * kD + d) * kS + s];
    red[tid] = p; __syncthreads();
    for (int w = 128; w > 0; w >>= 1) { if (tid < w) red[tid] += red[tid + w]; __syncthreads(); }
    if (tid == 0) {
        float m = red[0] * (1.0f / kD);
        sig[blockIdx.x] = 1.0f / (1.0f + __expf(-m));
    }
}

// xh = f16(gamma[b,d]*X[b,t,d] + beta[b,d]); vectorized x4
struct H4 { half_t a, b, c, d; };
__global__ void film_kernel(const float* __restrict__ X, const float* __restrict__ gamma,
                            const float* __restrict__ beta, half_t* __restrict__ xh) {
    int i4 = blockIdx.x * 256 + threadIdx.x;            // over M*D/4
    int i = i4 * 4;
    int b = i / (kT * kD);
    int d = i & (kD - 1);
    float4 xv = *(const float4*)(X + i);
    float4 gv = *(const float4*)(gamma + (size_t)b * kD + d);
    float4 bv = *(const float4*)(beta  + (size_t)b * kD + d);
    H4 o;
    o.a = (half_t)(gv.x * xv.x + bv.x);
    o.b = (half_t)(gv.y * xv.y + bv.y);
    o.c = (half_t)(gv.z * xv.z + bv.z);
    o.d = (half_t)(gv.w * xv.w + bv.w);
    *(H4*)(xh + i) = o;
}

// ---------------- WMMA GEMM: C(MxN) = A(MxK) * W(NxK)^T ---------------------
// Double-buffered LDS + single-tile register prefetch; one barrier per K-step.
// #pragma unroll 1 on the K loop keeps exactly one prefetch tile live (no spills).
// MODE 0: delta GEMM -> softplus + row-mean atomic into dm (no C store)
// MODE 1: Wo GEMM    -> out = scal[m]*acc + bias[n] + resid[m,n]; store f32
// MODE 2: BC GEMM    -> out = acc + bias[n]; store f32 (N = 128)
template <int MODE>
__global__ __launch_bounds__(256)
void gemm_kernel(const half_t* __restrict__ Aq, const half_t* __restrict__ Wq,
                 const float* __restrict__ bias, float* __restrict__ outF,
                 float* __restrict__ dmOut, const float* __restrict__ scal,
                 const half_t* __restrict__ resid, int Nld) {
    __shared__ half_t As[2][128][72];
    __shared__ half_t Bs[2][128][72];

    const int tid = threadIdx.x;
    const int lane = tid & 31;
    const int wave = tid >> 5;
    const int wm = wave & 3, wn = wave >> 2;      // 4 waves along M, 2 along N
    const int mW = wm * 32, nW = wn * 64;
    const int laneLo = lane & 15, halfSel = lane >> 4;
    const int mBlock = blockIdx.y * 128;
    const int nBlock = blockIdx.x * 128;

    v8f acc[2][4];
    const v8f vzero = {0.f, 0.f, 0.f, 0.f, 0.f, 0.f, 0.f, 0.f};
#pragma unroll
    for (int mi = 0; mi < 2; ++mi)
#pragma unroll
        for (int ni = 0; ni < 4; ++ni) acc[mi][ni] = vzero;

    const half_t* Ab = Aq + (size_t)mBlock * kD;
    const half_t* Wb = Wq + (size_t)nBlock * kD;
    const int lr = tid >> 3;
    const int lc = (tid & 7) * 8;                 // half offset within 64-wide K tile

    // prologue: tile 0 straight into LDS buffer 0
#pragma unroll
    for (int p = 0; p < 4; ++p) {
        int r = lr + p * 32;
        uint4 va = *(const uint4*)(Ab + (size_t)r * kD + lc);
        uint4 vb = *(const uint4*)(Wb + (size_t)r * kD + lc);
        *(uint4*)&As[0][r][lc] = va;
        *(uint4*)&Bs[0][r][lc] = vb;
    }
    __syncthreads();

    constexpr int NIT = kD / 64;                  // 16 K-steps
#pragma unroll 1
    for (int it = 0; it < NIT; ++it) {
        const int cur = it & 1;
        const bool pf = (it + 1 < NIT);

        uint4 ta[4], tb[4];                       // single-tile staging, loop-local
        if (pf) {
            const int k0 = (it + 1) * 64;
#pragma unroll
            for (int p = 0; p < 4; ++p) {
                int r = lr + p * 32;
                ta[p] = *(const uint4*)(Ab + (size_t)r * kD + k0 + lc);
                tb[p] = *(const uint4*)(Wb + (size_t)r * kD + k0 + lc);
            }
        }

#pragma unroll
        for (int kc = 0; kc < 2; ++kc) {
            const int kb = kc * 32;
            FragU aF[2], bF[4];
#pragma unroll
            for (int mi = 0; mi < 2; ++mi) {
                int r = mW + mi * 16 + laneLo;
                int c = kb + halfSel * 8;         // A: lanes0-15 K{0-7,16-23}, lanes16-31 K{8-15,24-31}
                aF[mi].u[0] = *(const uint4*)&As[cur][r][c];
                aF[mi].u[1] = *(const uint4*)&As[cur][r][c + 16];
            }
#pragma unroll
            for (int ni = 0; ni < 4; ++ni) {
                int r = nW + ni * 16 + laneLo;    // B: lane = N column, lanes0-15 K{0-7,16-23}, lanes16-31 K{8-15,24-31}
                int c = kb + halfSel * 8;
                bF[ni].u[0] = *(const uint4*)&Bs[cur][r][c];
                bF[ni].u[1] = *(const uint4*)&Bs[cur][r][c + 16];
            }
#pragma unroll
            for (int mi = 0; mi < 2; ++mi)
#pragma unroll
                for (int ni = 0; ni < 4; ++ni)
                    acc[mi][ni] = __builtin_amdgcn_wmma_f32_16x16x32_f16(
                        false, aF[mi].v, false, bF[ni].v, (short)0, acc[mi][ni], false, false);
        }

        if (pf) {
#pragma unroll
            for (int p = 0; p < 4; ++p) {
                int r = lr + p * 32;
                *(uint4*)&As[1 - cur][r][lc] = ta[p];
                *(uint4*)&Bs[1 - cur][r][lc] = tb[p];
            }
        }
        __syncthreads();
    }

    // ---------------- epilogue ----------------
    if (MODE == 0) {
#pragma unroll
        for (int mi = 0; mi < 2; ++mi) {
#pragma unroll
            for (int vi = 0; vi < 8; ++vi) {
                float rs = 0.f;
#pragma unroll
                for (int ni = 0; ni < 4; ++ni) {
                    int n = nBlock + nW + ni * 16 + laneLo;
                    rs += softplusf(acc[mi][ni][vi] + bias[n]);
                }
                rs += __shfl_xor(rs, 1);
                rs += __shfl_xor(rs, 2);
                rs += __shfl_xor(rs, 4);
                rs += __shfl_xor(rs, 8);
                if (laneLo == 0) {
                    int m = mBlock + mW + mi * 16 + vi + halfSel * 8;
                    atomicAdd(&dmOut[m], rs * (1.0f / kD));
                }
            }
        }
    } else {
#pragma unroll
        for (int mi = 0; mi < 2; ++mi)
#pragma unroll
            for (int ni = 0; ni < 4; ++ni)
#pragma unroll
                for (int vi = 0; vi < 8; ++vi) {
                    int m = mBlock + mW + mi * 16 + vi + halfSel * 8;
                    int n = nBlock + nW + ni * 16 + laneLo;
                    float v = acc[mi][ni][vi];
                    if (MODE == 1)
                        v = v * scal[m] + bias[n] + (float)resid[(size_t)m * kD + n];
                    else
                        v = v + bias[n];
                    outF[(size_t)m * Nld + n] = v;
                }
    }
}

// ---------------- chunked scan (h_t = a_t h + u_t), 3 passes ----------------
// 32 threads, each lane owns s=lane and s=lane+32.
__global__ void scan_pass1(const float* __restrict__ dm, const float* __restrict__ BtCt,
                           const float* __restrict__ sig, float* __restrict__ Uend,
                           float* __restrict__ Aprod) {
    int b = blockIdx.x / kNC, c = blockIdx.x % kNC;
    int s0 = threadIdx.x, s1 = s0 + 32;
    float sg0 = sig[s0], sg1 = sig[s1];
    float h0 = 0.f, h1 = 0.f, ap = 1.f;
    int t0 = c * kCH;
    for (int i = 0; i < kCH; ++i) {
        int row = b * kT + t0 + i;
        float d = dm[row], a = 1.f - d;
        const float* bc = BtCt + (size_t)row * 128;
        h0 = h0 * a + d * bc[s0] * sg0;
        h1 = h1 * a + d * bc[s1] * sg1;
        ap *= a;
    }
    Uend[(size_t)blockIdx.x * 64 + s0] = h0;
    Uend[(size_t)blockIdx.x * 64 + s1] = h1;
    if (threadIdx.x == 0) Aprod[blockIdx.x] = ap;
}

__global__ void scan_pass2(const float* __restrict__ Uend, const float* __restrict__ Aprod,
                           float* __restrict__ h0buf) {
    int b = blockIdx.x;
    int s0 = threadIdx.x, s1 = s0 + 32;
    float h0 = 0.f, h1 = 0.f;
    for (int c = 0; c < kNC; ++c) {
        size_t idx = ((size_t)b * kNC + c) * 64;
        h0buf[idx + s0] = h0;
        h0buf[idx + s1] = h1;
        float ap = Aprod[b * kNC + c];
        h0 = h0 * ap + Uend[idx + s0];
        h1 = h1 * ap + Uend[idx + s1];
    }
}

__global__ void scan_pass3(const float* __restrict__ dm, const float* __restrict__ BtCt,
                           const float* __restrict__ sig, const float* __restrict__ h0buf,
                           float* __restrict__ scal) {
    int b = blockIdx.x / kNC, c = blockIdx.x % kNC;
    int s0 = threadIdx.x, s1 = s0 + 32;
    float sg0 = sig[s0], sg1 = sig[s1];
    size_t idx = (size_t)blockIdx.x * 64;
    float h0 = h0buf[idx + s0], h1 = h0buf[idx + s1];
    int t0 = c * kCH;
    for (int i = 0; i < kCH; ++i) {
        int row = b * kT + t0 + i;
        float d = dm[row], a = 1.f - d;
        const float* bc = BtCt + (size_t)row * 128;
        h0 = h0 * a + d * bc[s0] * sg0;
        h1 = h1 * a + d * bc[s1] * sg1;
        float p = h0 * bc[64 + s0] + h1 * bc[64 + s1];
        p += __shfl_xor(p, 16);
        p += __shfl_xor(p, 8);
        p += __shfl_xor(p, 4);
        p += __shfl_xor(p, 2);
        p += __shfl_xor(p, 1);
        if (threadIdx.x == 0) scal[row] = p;
    }
}

// ---------------- head: out[b,a] = X[b,T-1,:] . Wfc[a,:] + bfc[a] ----------
__global__ void head_kernel(const float* __restrict__ X, const float* __restrict__ Wfc,
                            const float* __restrict__ bfc, float* __restrict__ out) {
    __shared__ float red[256];
    int b = blockIdx.x / kA, a = blockIdx.x % kA, tid = threadIdx.x;
    const float* xr = X + ((size_t)b * kT + (kT - 1)) * kD;
    float p = 0.f;
    for (int d = tid; d < kD; d += 256) p += xr[d] * Wfc[(size_t)a * kD + d];
    red[tid] = p; __syncthreads();
    for (int w = 128; w > 0; w >>= 1) { if (tid < w) red[tid] += red[tid + w]; __syncthreads(); }
    if (tid == 0) out[b * kA + a] = red[0] + bfc[a];
}

// ---------------- launch ----------------------------------------------------
extern "C" void kernel_launch(void* const* d_in, const int* in_sizes, int n_in,
                              void* d_out, int out_size, void* d_ws, size_t ws_size,
                              hipStream_t stream) {
    const float* x     = (const float*)d_in[0];
    const float* gamma = (const float*)d_in[1];
    const float* beta  = (const float*)d_in[2];
    const float* Wd    = (const float*)d_in[3];
    const float* bd    = (const float*)d_in[4];
    const float* WB    = (const float*)d_in[5];
    const float* bB    = (const float*)d_in[6];
    const float* WC    = (const float*)d_in[7];
    const float* bC    = (const float*)d_in[8];
    const float* Wo    = (const float*)d_in[9];
    const float* bo    = (const float*)d_in[10];
    const float* Am    = (const float*)d_in[11];
    const float* Wfc   = (const float*)d_in[12];
    const float* bfc   = (const float*)d_in[13];
    float* out = (float*)d_out;

    char* w = (char*)d_ws;
    auto alloc = [&](size_t bytes) -> char* {
        char* p = w; w += (bytes + 255) & ~size_t(255); return p;
    };
    float*  X     = (float*)alloc((size_t)kM * kD * 4);          // 64 MB
    half_t* xh    = (half_t*)alloc((size_t)kM * kD * 2);         // 32 MB
    half_t* Wdh   = (half_t*)alloc((size_t)kL * kD * kD * 2);    // 8 MB
    half_t* Woh   = (half_t*)alloc((size_t)kL * kD * kD * 2);    // 8 MB
    half_t* WBCh  = (half_t*)alloc((size_t)kL * 128 * kD * 2);   // 1 MB
    float*  bBC   = (float*)alloc(kL * 128 * 4);
    float*  BtCt  = (float*)alloc((size_t)kM * 128 * 4);         // 8 MB
    float*  dm    = (float*)alloc(kM * 4);
    float*  scal  = (float*)alloc(kM * 4);
    float*  sig   = (float*)alloc(kL * kS * 4);
    float*  Aprod = (float*)alloc(kB * kNC * 4);
    float*  Uend  = (float*)alloc((size_t)kB * kNC * 64 * 4);
    float*  h0buf = (float*)alloc((size_t)kB * kNC * 64 * 4);

    const int nW = kL * kD * kD;
    cvt_f16_kernel<<<nW / 256, 256, 0, stream>>>(Wd, Wdh, nW);
    cvt_f16_kernel<<<nW / 256, 256, 0, stream>>>(Wo, Woh, nW);
    pack_wbc_kernel<<<(kL * 128 * kD) / 256, 256, 0, stream>>>(WB, WC, WBCh);
    pack_bias_kernel<<<1, kL * 128, 0, stream>>>(bB, bC, bBC);
    sig_kernel<<<kL * kS, 256, 0, stream>>>(Am, sig);

    for (int l = 0; l < kL; ++l) {
        const float* xin = (l == 0) ? x : X;
        film_kernel<<<(kM * kD) / 1024, 256, 0, stream>>>(xin, gamma, beta, xh);

        hipMemsetAsync(dm, 0, kM * sizeof(float), stream);
        // delta GEMM (M x 1024 x 1024) -> fused softplus + row-mean into dm
        gemm_kernel<0><<<dim3(kD / 128, kM / 128), 256, 0, stream>>>(
            xh, Wdh + (size_t)l * kD * kD, bd + (size_t)l * kD,
            nullptr, dm, nullptr, nullptr, kD);
        // Bt|Ct GEMM (M x 128 x 1024)
        gemm_kernel<2><<<dim3(1, kM / 128), 256, 0, stream>>>(
            xh, WBCh + (size_t)l * 128 * kD, bBC + l * 128,
            BtCt, nullptr, nullptr, nullptr, 128);

        scan_pass1<<<kB * kNC, 32, 0, stream>>>(dm, BtCt, sig + l * kS, Uend, Aprod);
        scan_pass2<<<kB, 32, 0, stream>>>(Uend, Aprod, h0buf);
        scan_pass3<<<kB * kNC, 32, 0, stream>>>(dm, BtCt, sig + l * kS, h0buf, scal);

        // Wo GEMM (M x 1024 x 1024), epilogue: *scal[m] + bo + residual(xh) -> X
        gemm_kernel<1><<<dim3(kD / 128, kM / 128), 256, 0, stream>>>(
            xh, Woh + (size_t)l * kD * kD, bo + (size_t)l * kD,
            X, nullptr, scal, xh, kD);
    }

    head_kernel<<<kB * kA, 256, 0, stream>>>(X, Wfc, bfc, out);
}
